// SelfAttention_11149735100428
// MI455X (gfx1250) — matrix operand order repaired and checked by reading, not compiled, not run on hardware
//
#include <hip/hip_runtime.h>

// Problem constants (match reference)
#define BB  2
#define LL  2048
#define DD  1024
#define HH  16
#define KDD 64
#define E3  3072   // 3*D

typedef __attribute__((ext_vector_type(16))) __bf16 v16bf;
typedef __attribute__((ext_vector_type(8)))  __bf16 v8bf;
typedef __attribute__((ext_vector_type(2)))  __bf16 v2bf;
typedef __attribute__((ext_vector_type(8)))  float  v8f;

__device__ __forceinline__ v8f wmma_bf16(v16bf a, v16bf b, v8f c) {
  // D = A(16x32) * B(32x16) + C, f32 accumulate
  return __builtin_amdgcn_wmma_f32_16x16x32_bf16(false, a, false, b, (short)0, c, false, false);
}

// ---- DPP16 row-rotation reductions (16-lane rows, no LDS, no waits) --------
template <int CTRL>
__device__ __forceinline__ float dpp_rot(float x) {
  return __int_as_float(
      __builtin_amdgcn_update_dpp(0, __float_as_int(x), CTRL, 0xf, 0xf, true));
}
__device__ __forceinline__ float row_max16(float x) {
  x = fmaxf(x, dpp_rot<0x121>(x));   // row_ror:1
  x = fmaxf(x, dpp_rot<0x122>(x));   // row_ror:2
  x = fmaxf(x, dpp_rot<0x124>(x));   // row_ror:4
  x = fmaxf(x, dpp_rot<0x128>(x));   // row_ror:8
  return x;
}
__device__ __forceinline__ float row_sum16(float x) {
  x += dpp_rot<0x121>(x);
  x += dpp_rot<0x122>(x);
  x += dpp_rot<0x124>(x);
  x += dpp_rot<0x128>(x);
  return x;
}

// Assemble a 16-element bf16 fragment from two 16B-aligned 8-element runs.
__device__ __forceinline__ v16bf join8(v8bf lo, v8bf hi) {
  return __builtin_shufflevector(lo, hi, 0, 1, 2, 3, 4, 5, 6, 7,
                                         8, 9, 10, 11, 12, 13, 14, 15);
}
__device__ __forceinline__ v8bf ld8(const __bf16* p) { return *(const v8bf*)p; }

// Pack two floats into adjacent bf16 slots (one cvt_pk + one b32 store).
__device__ __forceinline__ void st_pk(__bf16* p, float a, float b) {
  v2bf t; t[0] = (__bf16)a; t[1] = (__bf16)b;
  *(v2bf*)p = t;
}

// ---------------------------------------------------------------------------
// GEMM: C(MxN) = A(MxK) * W(KxN); A is f32 or bf16, C is f32 or bf16,
// W is f32 (model weights). bf16 WMMA math, f32 accumulate.
// Block: 256 threads (8 waves, 2x4 grid), block tile 64(M) x 256(N), K-step 64.
// Each wave owns a 32x64 tile (2x4 WMMA tiles): 16 WMMAs per LDS stage.
// W tile stored column-major in LDS (k-pairs packed as b32) so B-fragments
// are contiguous b128 loads.
// ---------------------------------------------------------------------------
template <typename TA, typename TC, int N, int K>
__global__ __launch_bounds__(256)
void gemm_bf16w(const TA* __restrict__ A, const float* __restrict__ W,
                TC* __restrict__ C) {
  __shared__ __bf16 As[64][72];    // [m][k]  row stride 144B (16B multiple)
  __shared__ __bf16 Bt[256][72];   // [n][k]  row stride 144B (16B multiple)

  const int tid  = threadIdx.x;
  const int lane = tid & 31;
  const int wid  = tid >> 5;
  const int ln   = lane & 15;
  const int g    = lane >> 4;

  const int m0 = blockIdx.y * 64;
  const int n0 = blockIdx.x * 256;
  const int rb = (wid >> 2) * 32;            // wave row base in block tile
  const int cb = (wid & 3) * 64;             // wave col base in block tile

  const int ar = tid >> 2, ac = (tid & 3) * 16;   // A fill: 16 elems/thread
  const int wk = (tid & 31) * 2;                  // W fill: k-pair base
  const int wn = (tid >> 5) * 32;                 // W fill: 32 n/thread

  v8f acc[2][4] = {};

  for (int k0 = 0; k0 < K; k0 += 64) {
    if (k0 + 64 < K) {  // prefetch next K-slab while this stage computes
      __builtin_prefetch(&A[(size_t)(m0 + ar) * K + k0 + 64 + ac], 0, 3);
      __builtin_prefetch(&W[(size_t)(k0 + 64 + wk) * N + n0 + wn], 0, 3);
    }
    // A tile: row-major; f32 source converts (packed), bf16 source raw copy
    if constexpr (sizeof(TA) == 4) {
      const float4* ap = (const float4*)&A[(size_t)(m0 + ar) * K + k0 + ac];
      float4 t0 = ap[0], t1 = ap[1], t2 = ap[2], t3 = ap[3];
      v8bf lo, hi;
      lo[0] = (__bf16)t0.x; lo[1] = (__bf16)t0.y; lo[2] = (__bf16)t0.z; lo[3] = (__bf16)t0.w;
      lo[4] = (__bf16)t1.x; lo[5] = (__bf16)t1.y; lo[6] = (__bf16)t1.z; lo[7] = (__bf16)t1.w;
      hi[0] = (__bf16)t2.x; hi[1] = (__bf16)t2.y; hi[2] = (__bf16)t2.z; hi[3] = (__bf16)t2.w;
      hi[4] = (__bf16)t3.x; hi[5] = (__bf16)t3.y; hi[6] = (__bf16)t3.z; hi[7] = (__bf16)t3.w;
      *(v8bf*)&As[ar][ac]     = lo;
      *(v8bf*)&As[ar][ac + 8] = hi;
    } else {
      const v8bf* ap = (const v8bf*)&A[(size_t)(m0 + ar) * K + k0 + ac];
      *(v8bf*)&As[ar][ac]     = ap[0];
      *(v8bf*)&As[ar][ac + 8] = ap[1];
    }
    // W tile: two k-rows read (float4), transposed writes as packed k-pairs
    {
      const float4* b0 = (const float4*)&W[(size_t)(k0 + wk)     * N + n0 + wn];
      const float4* b1 = (const float4*)&W[(size_t)(k0 + wk + 1) * N + n0 + wn];
      #pragma unroll
      for (int i = 0; i < 8; ++i) {
        float4 r0 = b0[i], r1 = b1[i];
        st_pk(&Bt[wn + 4 * i + 0][wk], r0.x, r1.x);
        st_pk(&Bt[wn + 4 * i + 1][wk], r0.y, r1.y);
        st_pk(&Bt[wn + 4 * i + 2][wk], r0.z, r1.z);
        st_pk(&Bt[wn + 4 * i + 3][wk], r0.w, r1.w);
      }
    }
    __syncthreads();

    #pragma unroll
    for (int kh = 0; kh < 2; ++kh) {
      v16bf af[2], bfr[4];
      #pragma unroll
      for (int i = 0; i < 2; ++i) {
        const int row = rb + i * 16 + ln;
        // ISA A layout per lane = two contiguous 8-element runs
        af[i] = join8(*(const v8bf*)&As[row][kh * 32 + g * 8],
                      *(const v8bf*)&As[row][kh * 32 + 16 + g * 8]);
      }
      #pragma unroll
      for (int j = 0; j < 4; ++j) {
        const int col = cb + j * 16 + ln;
        // ISA B layout per lane = two contiguous 8-element runs (same k order as A)
        bfr[j] = join8(*(const v8bf*)&Bt[col][kh * 32 + g * 8],
                       *(const v8bf*)&Bt[col][kh * 32 + 16 + g * 8]);
      }
      #pragma unroll
      for (int i = 0; i < 2; ++i)
        #pragma unroll
        for (int j = 0; j < 4; ++j)
          acc[i][j] = wmma_bf16(af[i], bfr[j], acc[i][j]);
    }
    __syncthreads();
  }

  #pragma unroll
  for (int i = 0; i < 2; ++i)
    #pragma unroll
    for (int j = 0; j < 4; ++j)
      #pragma unroll
      for (int r = 0; r < 8; ++r) {                 // ISA C layout
        const int m = m0 + rb + i * 16 + r + 8 * g;
        const int n = n0 + cb + j * 16 + ln;
        C[(size_t)m * N + n] = (TC)acc[i][j][r];
      }
}

// ---------------------------------------------------------------------------
// Pointer into bf16 qkv implementing the reference's "faithful" reshape:
//   qh[b,h,l,kd] = q_flat[b][h*L*KD + l*KD + kd], q/k/v interleaved in the
//   (B, L, 3D) qkv tensor. sel: 0=q, 1=k, 2=v. Spans used are row-contiguous.
// ---------------------------------------------------------------------------
__device__ __forceinline__ const __bf16* qkv_ptr(const __bf16* __restrict__ qkv,
                                                 int b, int h, int l, int kd, int sel) {
  const int flat = h * (LL * KDD) + l * KDD + kd;
  return qkv + ((size_t)b * LL + (flat >> 10)) * E3 + sel * DD + (flat & (DD - 1));
}

// ---------------------------------------------------------------------------
// Flash attention: one wave (32 threads) per (b, h, 32-query tile).
// qkv is bf16: fragments load directly (zero conversion VALU in the K/V path).
// Two 16-row M-tiles (u=0,1) share K fragments and the V LDS tile. Online
// softmax runs in the exp2 domain (scale 1/8 folded into log2e) with DPP row
// reductions; v_exp_f32 is natively exp2 so no hidden multiplies remain.
// ---------------------------------------------------------------------------
__global__ __launch_bounds__(32)
void attn_flash(const __bf16* __restrict__ qkv, __bf16* __restrict__ att) {
  __shared__ __bf16 Plds[32][40];   // probs C-layout -> A-layout (stride 80B)
  __shared__ __bf16 Vlds[64][40];   // [kd][s] transposed V chunk (stride 80B)

  const int lane = threadIdx.x & 31;
  const int ln   = lane & 15;
  const int g    = lane >> 4;
  const int q0   = blockIdx.x * 32;
  const int h    = blockIdx.y;
  const int b    = blockIdx.z;

  // Q fragments: 32 queries x 64 kd -> 2 M-tiles x 2 K-slabs of 16x32 bf16.
  v16bf qf[2][2];
  #pragma unroll
  for (int u = 0; u < 2; ++u)
    #pragma unroll
    for (int kdh = 0; kdh < 2; ++kdh)
      qf[u][kdh] = join8(
          ld8(qkv_ptr(qkv, b, h, q0 + u * 16 + ln, kdh * 32 + g * 8, 0)),
          ld8(qkv_ptr(qkv, b, h, q0 + u * 16 + ln, kdh * 32 + 16 + g * 8, 0)));

  const float NEG    = -1e30f;
  const float CSCALE = 0.125f * 1.44269504089f;  // (1/sqrt(64)) * log2(e)
  float mrow[2][8], lsum[2][8];
  v8f oacc[2][4] = {};
  #pragma unroll
  for (int u = 0; u < 2; ++u)
    #pragma unroll
    for (int r = 0; r < 8; ++r) { mrow[u][r] = NEG; lsum[u][r] = 0.f; }

  for (int s0 = 0; s0 <= q0 + 31; s0 += 32) {
    // S = Q * K^T: K fragments loaded once (pure b128 loads), both M-tiles.
    v8f sacc[2][2] = {};
    #pragma unroll
    for (int kdh = 0; kdh < 2; ++kdh) {
      const __bf16* p0 = qkv_ptr(qkv, b, h, s0 + ln,      kdh * 32 + g * 8, 1);
      const __bf16* p1 = qkv_ptr(qkv, b, h, s0 + 16 + ln, kdh * 32 + g * 8, 1);
      const v16bf kf0 = join8(ld8(p0), ld8(qkv_ptr(qkv, b, h, s0 + ln,      kdh * 32 + 16 + g * 8, 1)));
      const v16bf kf1 = join8(ld8(p1), ld8(qkv_ptr(qkv, b, h, s0 + 16 + ln, kdh * 32 + 16 + g * 8, 1)));
      #pragma unroll
      for (int u = 0; u < 2; ++u) {
        sacc[u][0] = wmma_bf16(qf[u][kdh], kf0, sacc[u][0]);
        sacc[u][1] = wmma_bf16(qf[u][kdh], kf1, sacc[u][1]);
      }
    }

    // Online softmax (exp2 domain). Lane (ln, g) holds rows r+8g, col ln;
    // reductions are DPP row-rotations confined to each 16-lane half.
    float p0v[2][8], p1v[2][8];
    #pragma unroll
    for (int u = 0; u < 2; ++u)
      #pragma unroll
      for (int r = 0; r < 8; ++r) {
        const int q  = q0 + u * 16 + r + 8 * g;
        const int sa = s0 + ln, sb = sa + 16;
        float v0 = sacc[u][0][r] * CSCALE; if (sa > q) v0 = NEG;  // causal
        float v1 = sacc[u][1][r] * CSCALE; if (sb > q) v1 = NEG;
        const float rm   = row_max16(fmaxf(v0, v1));
        const float nm   = fmaxf(mrow[u][r], rm);
        const float corr = exp2f(mrow[u][r] - nm);
        mrow[u][r] = nm;
        const float e0 = exp2f(v0 - nm);
        const float e1 = exp2f(v1 - nm);
        lsum[u][r] = lsum[u][r] * corr + row_sum16(e0 + e1);
        #pragma unroll
        for (int j = 0; j < 4; ++j) oacc[u][j][r] *= corr;
        p0v[u][r] = e0; p1v[u][r] = e1;
      }

    __syncthreads();  // WAR: previous iteration's Plds/Vlds reads done
    // P: C-layout -> A-layout staging (both M-tiles)
    #pragma unroll
    for (int u = 0; u < 2; ++u)
      #pragma unroll
      for (int r = 0; r < 8; ++r) {
        Plds[u * 16 + r + 8 * g][ln]      = (__bf16)p0v[u][r];
        Plds[u * 16 + r + 8 * g][16 + ln] = (__bf16)p1v[u][r];
      }
    // V chunk: each lane streams one s-row (64 consecutive bf16, b128 loads),
    // writes transposed so fragment reads are contiguous. No conversion.
    {
      const v8bf* vp = (const v8bf*)qkv_ptr(qkv, b, h, s0 + lane, 0, 2);
      #pragma unroll
      for (int i = 0; i < 8; ++i) {
        v8bf t = vp[i];
        #pragma unroll
        for (int e = 0; e < 8; ++e)
          Vlds[8 * i + e][lane] = t[e];
      }
    }
    __syncthreads();

    v16bf pf[2];
    #pragma unroll
    for (int u = 0; u < 2; ++u)    // A layout: two contiguous 8-element runs
      pf[u] = join8(*(const v8bf*)&Plds[u * 16 + ln][g * 8],
                    *(const v8bf*)&Plds[u * 16 + ln][16 + g * 8]);
    // O += P(16x32) * V(32x64): V fragment loaded once per j, used twice.
    #pragma unroll
    for (int j = 0; j < 4; ++j) {
      const v16bf vf = join8(*(const v8bf*)&Vlds[j * 16 + ln][g * 8],
                             *(const v8bf*)&Vlds[j * 16 + ln][16 + g * 8]);
      #pragma unroll
      for (int u = 0; u < 2; ++u)
        oacc[u][j] = wmma_bf16(pf[u], vf, oacc[u][j]);
    }
    __syncthreads();
  }

  // Normalize (v_rcp_f32 + mul instead of full divide) and emit bf16 in
  // standard concat layout: att[b, q, h*64 + kd]
  #pragma unroll
  for (int u = 0; u < 2; ++u) {
    float inv[8];
    #pragma unroll
    for (int r = 0; r < 8; ++r) inv[r] = __builtin_amdgcn_rcpf(lsum[u][r]);
    #pragma unroll
    for (int j = 0; j < 4; ++j)
      #pragma unroll
      for (int r = 0; r < 8; ++r) {
        const int q = q0 + u * 16 + r + 8 * g;
        att[((size_t)b * LL + q) * DD + h * KDD + j * 16 + ln] =
            (__bf16)(oacc[u][j][r] * inv[r]);
      }
  }
}

// ---------------------------------------------------------------------------
extern "C" void kernel_launch(void* const* d_in, const int* in_sizes, int n_in,
                              void* d_out, int out_size, void* d_ws, size_t ws_size,
                              hipStream_t stream) {
  (void)in_sizes; (void)n_in; (void)out_size; (void)ws_size;
  const float* x    = (const float*)d_in[0];   // (B, L, D)
  const float* wqkv = (const float*)d_in[1];   // (D, 3D)
  const float* wend = (const float*)d_in[2];   // (D, D)
  float* out = (float*)d_out;                  // (B, L, D) f32

  __bf16* qkvb = (__bf16*)d_ws;                    // (B*L, 3D) bf16 : 25 MB
  __bf16* attb = qkvb + (size_t)BB * LL * E3;      // (B*L, D)  bf16 :  8 MB

  const int M = BB * LL;  // 4096 token rows

  // 1) QKV projection: f32 in, bf16 out (single conversion at the producer)
  gemm_bf16w<float, __bf16, E3, DD>
      <<<dim3(E3 / 256, M / 64), dim3(256), 0, stream>>>(x, wqkv, qkvb);
  // 2) Causal flash attention per (b, h, 32-query tile), bf16 in/out
  attn_flash<<<dim3(LL / 32, HH, BB), dim3(32), 0, stream>>>(qkvb, attb);
  // 3) Output projection: bf16 in, f32 out
  gemm_bf16w<__bf16, float, DD, DD>
      <<<dim3(DD / 256, M / 64), dim3(256), 0, stream>>>(attb, wend, out);
}
